// Attention_1185410974409
// MI455X (gfx1250) — matrix operand order repaired and checked by reading, not compiled, not run on hardware
//
#include <hip/hip_runtime.h>
#include <hip/hip_bf16.h>

typedef __attribute__((ext_vector_type(16))) _Float16 v16h;
typedef __attribute__((ext_vector_type(8)))  _Float16 v8h;
typedef __attribute__((ext_vector_type(4)))  _Float16 v4h;
typedef __attribute__((ext_vector_type(2)))  _Float16 v2h;
typedef __attribute__((ext_vector_type(8)))  float    v8f;

// Problem constants (match reference)
#define DX    1024
#define DZ    1024
#define DATTN 1024
#define DOUT  1024
#define LX    4096
#define LZ    4096

// Tiling
#define BM 128
#define BN 128
#define BK 32
#define LDSK (BK + 8)   // padded stride in halves (80B rows; 80 % 16 == 0 -> aligned b128 sub-loads)

// gfx1250 async global->LDS DMA for f16 A-tiles (straight copy, no convert/transpose)
#define USE_ASYNC_COPY 1

// Epilogue modes
#define EPI_F16   0   // out f16 [m*N+n], + bias[m]
#define EPI_F16T  1   // out f16 [n*M+m] (transposed), + bias[m]
#define EPI_MASK  2   // mask -> -1000 (before scale), *scale, out f32 [m*N+n]
#define EPI_F32   3   // out f32 [m*N+n]

__device__ __forceinline__ void async_wait_all() {
#if USE_ASYNC_COPY
    asm volatile("s_wait_asynccnt 0" ::: "memory");
#endif
}

template <bool AF32, bool BF32, int EPI>
__global__ __launch_bounds__(256, 2)
void wmma_gemm_kernel(const void* __restrict__ Ap, const void* __restrict__ Bp,
                      const float* __restrict__ bias,
                      const unsigned char* __restrict__ mask,
                      void* __restrict__ outp,
                      int M, int N, int K, float scale)
{
    // Double-buffered tiles: 2 * (128*40 + 128*40) * 2B = 40 KB (of 320 KB/WGP)
    __shared__ __align__(16) _Float16 As[2][BM][LDSK];   // [m][k]
    __shared__ __align__(16) _Float16 Bs[2][BN][LDSK];   // transposed: [n][k]

    const int tid   = threadIdx.x;
    const int lane  = tid & 31;
    const int wave  = tid >> 5;
    const int wr    = wave >> 1;       // 0..3 -> 32 rows each
    const int wc    = wave & 1;        // 0..1 -> 64 cols each
    const int m0    = blockIdx.y * BM;
    const int n0    = blockIdx.x * BN;
    const int lhalf = lane >> 4;       // 0 or 1
    const int lmod  = lane & 15;

    v8f acc[2][4];
#pragma unroll
    for (int tm = 0; tm < 2; ++tm)
#pragma unroll
        for (int tn = 0; tn < 4; ++tn)
#pragma unroll
            for (int r = 0; r < 8; ++r) acc[tm][tn][r] = 0.0f;

    // ---- staging registers (global -> regs -> LDS, split so loads get a full
    //      compute phase of latency cover instead of per-load s_wait_loadcnt 0)
    float    sa0[8], sa1[8];   // A f32 pairs
    v8h      ar[2];            // A f16 segments (non-async fallback)
    float    sb0[8], sb1[8];   // B f32 pairs
    _Float16 hb0[8], hb1[8];   // B f16 pairs

    // Issue global loads for chunk k0; async A-path DMAs straight into LDS buf `ba`.
    auto stage_load = [&](int k0, int ba) {
        if constexpr (AF32) {
#pragma unroll
            for (int i = 0; i < 8; ++i) {
                int pidx = i * 256 + tid;
                int r = pidx >> 4, k = (pidx & 15) * 2;
                const float* src = (const float*)Ap + (size_t)(m0 + r) * K + (k0 + k);
                sa0[i] = src[0]; sa1[i] = src[1];
            }
        } else {
#pragma unroll
            for (int i = 0; i < 2; ++i) {
                int seg = i * 256 + tid;             // 512 x 16B segments
                int r = seg >> 2, s = (seg & 3) * 8;
                const _Float16* src = (const _Float16*)Ap + (size_t)(m0 + r) * K + (k0 + s);
#if USE_ASYNC_COPY
                _Float16* dst = &As[ba][r][s];
                // gfx1250 async DMA: LDS byte address == low 32 bits of generic ptr
                unsigned           ldsa = (unsigned)(unsigned long long)(uintptr_t)dst;
                unsigned long long ga   = (unsigned long long)(uintptr_t)src;
                asm volatile("global_load_async_to_lds_b128 %0, %1, off"
                             :: "v"(ldsa), "v"(ga) : "memory");
#else
                ar[i] = *(const v8h*)src;
#endif
            }
        }
#pragma unroll
        for (int i = 0; i < 8; ++i) {
            int idx = i * 256 + tid;
            int n = idx & 127, k = (idx >> 7) * 2;
            size_t g0 = (size_t)(k0 + k) * N + (n0 + n);
            if constexpr (BF32) {
                sb0[i] = ((const float*)Bp)[g0];
                sb1[i] = ((const float*)Bp)[g0 + N];
            } else {
                hb0[i] = ((const _Float16*)Bp)[g0];
                hb1[i] = ((const _Float16*)Bp)[g0 + N];
            }
        }
        // Prefetch the chunk after next (beyond the double buffer) into L2
        if (k0 + BK < K) {
            const char* pa = (const char*)Ap +
                ((size_t)(m0 + (tid >> 1)) * K + (k0 + BK) + (tid & 1) * 16) * (AF32 ? 4 : 2);
            __builtin_prefetch(pa, 0, 1);
            const char* pb = (const char*)Bp +
                ((size_t)(k0 + BK + (tid >> 3)) * N + n0 + (tid & 7) * 16) * (BF32 ? 4 : 2);
            __builtin_prefetch(pb, 0, 1);
        }
    };

    // Convert/pack and store staged registers into LDS buffer b.
    auto stage_store = [&](int b) {
        if constexpr (AF32) {
#pragma unroll
            for (int i = 0; i < 8; ++i) {
                int pidx = i * 256 + tid;
                int r = pidx >> 4, k = (pidx & 15) * 2;
                v2h pk; pk[0] = (_Float16)sa0[i]; pk[1] = (_Float16)sa1[i];
                *(v2h*)&As[b][r][k] = pk;            // ds_store_b32
            }
        } else {
#if !USE_ASYNC_COPY
#pragma unroll
            for (int i = 0; i < 2; ++i) {
                int seg = i * 256 + tid;
                int r = seg >> 2, s = (seg & 3) * 8;
                *(v8h*)&As[b][r][s] = ar[i];
            }
#endif
        }
#pragma unroll
        for (int i = 0; i < 8; ++i) {
            int idx = i * 256 + tid;
            int n = idx & 127, k = (idx >> 7) * 2;
            v2h pk;
            if constexpr (BF32) { pk[0] = (_Float16)sb0[i]; pk[1] = (_Float16)sb1[i]; }
            else                { pk[0] = hb0[i];           pk[1] = hb1[i]; }
            *(v2h*)&Bs[b][n][k] = pk;                // ds_store_b32
        }
    };

    const int ksteps = K / BK;
    stage_load(0, 0);

    for (int kt = 0; kt < ksteps; ++kt) {
        const int b = kt & 1;
        stage_store(b);                          // regs (loaded last iter) -> buf b
        if constexpr (!AF32) async_wait_all();   // async DMAs into buf b must land
        __syncthreads();
        if (kt + 1 < ksteps) stage_load((kt + 1) * BK, b ^ 1);  // overlap with WMMAs

        // ---- fragments per ISA 7.12.2 lane layouts (wave32)
        v16h aF[2], bF[4];
#pragma unroll
        for (int tm = 0; tm < 2; ++tm) {
            int row = wr * 32 + tm * 16 + lmod;
            // A 16x32 f16: lane m%16; K = {8h..8h+7} then {16+8h..23+8h}
            v8h lo = *(const v8h*)&As[b][row][8 * lhalf];
            v8h hi = *(const v8h*)&As[b][row][16 + 8 * lhalf];
            aF[tm] = __builtin_shufflevector(lo, hi, 0,1,2,3,4,5,6,7,8,9,10,11,12,13,14,15);
        }
#pragma unroll
        for (int tn = 0; tn < 4; ++tn) {
            int col = wc * 64 + tn * 16 + lmod;
            // B 32x16 f16: lane n%16; K = {8h..8h+7} then {16+8h..23+8h} (LDS is [n][k])
            v8h lo = *(const v8h*)&Bs[b][col][8 * lhalf];
            v8h hi = *(const v8h*)&Bs[b][col][16 + 8 * lhalf];
            bF[tn] = __builtin_shufflevector(lo, hi, 0,1,2,3,4,5,6,7,8,9,10,11,12,13,14,15);
        }

#pragma unroll
        for (int tm = 0; tm < 2; ++tm)
#pragma unroll
            for (int tn = 0; tn < 4; ++tn)
                acc[tm][tn] = __builtin_amdgcn_wmma_f32_16x16x32_f16(
                    /*neg_a=*/false, aF[tm], /*neg_b=*/false, bF[tn],
                    /*c_mod=*/(short)0, acc[tm][tn],
                    /*reuse_a=*/false, /*reuse_b=*/false);
    }

    // ---- epilogue: C/D layout m = r + 8*lhalf, n = lmod
#pragma unroll
    for (int tm = 0; tm < 2; ++tm)
#pragma unroll
        for (int tn = 0; tn < 4; ++tn)
#pragma unroll
            for (int r = 0; r < 8; ++r) {
                int mg = m0 + wr * 32 + tm * 16 + r + 8 * lhalf;
                int ng = n0 + wc * 64 + tn * 16 + lmod;
                float v = acc[tm][tn][r];
                if constexpr (EPI == EPI_F16) {
                    v += bias[mg];
                    ((_Float16*)outp)[(size_t)mg * N + ng] = (_Float16)v;
                } else if constexpr (EPI == EPI_F16T) {
                    v += bias[mg];
                    ((_Float16*)outp)[(size_t)ng * M + mg] = (_Float16)v;
                } else if constexpr (EPI == EPI_MASK) {
                    if (!mask[(size_t)mg * N + ng]) v = -1000.0f;  // mask BEFORE scale
                    v *= scale;
                    ((float*)outp)[(size_t)mg * N + ng] = v;
                } else {
                    ((float*)outp)[(size_t)mg * N + ng] = v;
                }
            }
}

// Softmax pass 1: per-column online (max, sum) with 4 row-slab partials per column.
// Block = 256 threads = 64 columns x 4 slabs; LDS merge; writes Mx[LX], Sum[LX].
__global__ __launch_bounds__(256)
void softmax_reduce_kernel(const float* __restrict__ S,
                           float* __restrict__ Mx, float* __restrict__ Sum)
{
    __shared__ float sm[4][64];
    __shared__ float sl[4][64];
    const int c  = threadIdx.x & 63;         // column within block
    const int t  = threadIdx.x >> 6;         // row-slab 0..3
    const int lx = blockIdx.x * 64 + c;

    float m = -3.402823e38f, l = 0.0f;
    const int lz0 = t * (LZ / 4);
    for (int i = 0; i < LZ / 4; ++i) {
        float s = S[(size_t)(lz0 + i) * LX + lx];
        float nm = fmaxf(m, s);
        l = l * __expf(m - nm) + __expf(s - nm);
        m = nm;
    }
    sm[t][c] = m; sl[t][c] = l;
    __syncthreads();
    if (t == 0) {
        float M = sm[0][c], L = sl[0][c];
#pragma unroll
        for (int j = 1; j < 4; ++j) {
            float mj = sm[j][c], lj = sl[j][c];
            float nm = fmaxf(M, mj);
            L = L * __expf(M - nm) + lj * __expf(mj - nm);
            M = nm;
        }
        Mx[lx] = M;
        Sum[lx] = 1.0f / L;
    }
}

// Softmax pass 2: fully parallel normalize; float4 reads, packed f16 writes.
__global__ __launch_bounds__(256)
void softmax_norm_kernel(const float* __restrict__ S,
                         const float* __restrict__ Mx, const float* __restrict__ Sum,
                         _Float16* __restrict__ P)
{
    size_t base = ((size_t)blockIdx.x * 256 + threadIdx.x) * 4;
    int lx0 = (int)(base & (LX - 1));
    float4 s = *(const float4*)(S + base);
    v4h o;
    o[0] = (_Float16)(__expf(s.x - Mx[lx0 + 0]) * Sum[lx0 + 0]);
    o[1] = (_Float16)(__expf(s.y - Mx[lx0 + 1]) * Sum[lx0 + 1]);
    o[2] = (_Float16)(__expf(s.z - Mx[lx0 + 2]) * Sum[lx0 + 2]);
    o[3] = (_Float16)(__expf(s.w - Mx[lx0 + 3]) * Sum[lx0 + 3]);
    *(v4h*)(P + base) = o;
}

extern "C" void kernel_launch(void* const* d_in, const int* in_sizes, int n_in,
                              void* d_out, int out_size, void* d_ws, size_t ws_size,
                              hipStream_t stream)
{
    const float* X  = (const float*)d_in[0];            // (DX, LX)
    const float* Z  = (const float*)d_in[1];            // (DZ, LZ)
    const unsigned char* mask = (const unsigned char*)d_in[2]; // (LZ, LX) bool (1B)
    const float* Wq = (const float*)d_in[3];            // (DATTN, DX)
    const float* bq = (const float*)d_in[4];            // (DATTN, 1)
    const float* Wk = (const float*)d_in[5];            // (DATTN, DZ)
    const float* bk = (const float*)d_in[6];
    const float* Wv = (const float*)d_in[7];            // (DOUT, DZ)
    const float* bv = (const float*)d_in[8];
    float* out = (float*)d_out;                         // (DOUT, LX)

    char* ws = (char*)d_ws;
    _Float16* q16  = (_Float16*)(ws);                               // DATTN*LX  f16 =  8 MiB
    _Float16* kT16 = (_Float16*)(ws + ((size_t)8  << 20));          // LZ*DATTN  f16 =  8 MiB
    _Float16* v16  = (_Float16*)(ws + ((size_t)16 << 20));          // DOUT*LZ   f16 =  8 MiB
    float*    Sc   = (float*)   (ws + ((size_t)24 << 20));          // LZ*LX     f32 = 64 MiB (L2-resident)
    _Float16* P16  = (_Float16*)(ws + ((size_t)88 << 20));          // LZ*LX     f16 = 32 MiB
    float*    Mx   = (float*)   (ws + ((size_t)120 << 20));         // LX f32
    float*    Sum  = (float*)   (ws + ((size_t)120 << 20) + (LX * 4)); // LX f32

    dim3 blk(256);
    const float inv_sqrt_dattn = 0.03125f;  // 1/sqrt(1024)

    // q = Wq @ X + bq  -> f16 (DATTN, LX)
    wmma_gemm_kernel<true, true, EPI_F16><<<dim3(LX / BN, DATTN / BM), blk, 0, stream>>>(
        Wq, X, bq, nullptr, q16, DATTN, LX, DX, 1.0f);
    // k = Wk @ Z + bk  -> f16 transposed (LZ, DATTN)
    wmma_gemm_kernel<true, true, EPI_F16T><<<dim3(LZ / BN, DATTN / BM), blk, 0, stream>>>(
        Wk, Z, bk, nullptr, kT16, DATTN, LZ, DZ, 1.0f);
    // v = Wv @ Z + bv  -> f16 (DOUT, LZ)
    wmma_gemm_kernel<true, true, EPI_F16><<<dim3(LZ / BN, DOUT / BM), blk, 0, stream>>>(
        Wv, Z, bv, nullptr, v16, DOUT, LZ, DZ, 1.0f);
    // score = kT @ q, mask -> -1000, * 1/32  -> f32 (LZ, LX)
    wmma_gemm_kernel<false, false, EPI_MASK><<<dim3(LX / BN, LZ / BM), blk, 0, stream>>>(
        kT16, q16, nullptr, mask, Sc, LZ, LX, DATTN, inv_sqrt_dattn);
    // softmax over lz (two passes)
    softmax_reduce_kernel<<<LX / 64, blk, 0, stream>>>(Sc, Mx, Sum);
    softmax_norm_kernel<<<(LZ * (size_t)LX) / (256 * 4), blk, 0, stream>>>(Sc, Mx, Sum, P16);
    // out = v @ attn  -> f32 (DOUT, LX)
    wmma_gemm_kernel<false, false, EPI_F32><<<dim3(LX / BN, DOUT / BM), blk, 0, stream>>>(
        v16, P16, nullptr, nullptr, out, DOUT, LX, LZ, 1.0f);
}
